// Attention_3702261809537
// MI455X (gfx1250) — matrix operand order repaired and checked by reading, not compiled, not run on hardware
//
#include <hip/hip_runtime.h>

// ---------------------------------------------------------------------------
// MI455X (gfx1250) multi-head attention, bf16 WMMA (v_wmma_f32_16x16x32_bf16)
// B=4, T=2048, D=1024, H=16, Dh=64.  Wave32.
// All heavy kernels use double-buffered GLOBAL_LOAD_ASYNC_TO_LDS_B128
// pipelines with ASYNCcnt waits; V is produced transposed [B][H][Dh][T].
// ---------------------------------------------------------------------------

#define CB 4
#define CT 2048
#define CD 1024
#define CH 16
#define CDH 64
#define KTILES (CD / 32)    // 32 k-steps of 32 (GEMMs)
#define STILES (CT / 32)    // 64 s-steps of 32 (attention)

typedef __attribute__((ext_vector_type(16))) __bf16 v16bf;
typedef __attribute__((ext_vector_type(8)))  float  v8f;

union Frag16 {
    v16bf v;
    unsigned short u[16];
    unsigned int   d[8];
};
union Acc8 {
    v8f   v;
    float f[8];
};

__device__ __forceinline__ unsigned short f2bf(float x) {
    unsigned int u = __float_as_uint(x);
    unsigned int r = u + 0x7FFFu + ((u >> 16) & 1u);   // round-to-nearest-even
    return (unsigned short)(r >> 16);
}

// A-matrix 16x32 bf16 pair layout (ISA 7.12.2): dword j of lane -> K of first elem
__device__ __forceinline__ int a_pair_k(int j, int half) {
    return (j < 4) ? (half * 8 + 2 * j) : (16 + half * 8 + 2 * (j - 4));
}

// Async DMA: global (16B) -> LDS, tracked by ASYNCcnt.
__device__ __forceinline__ void async_b128(unsigned lds_addr, const void* gaddr) {
    asm volatile("global_load_async_to_lds_b128 %0, %1, off"
                 :: "v"(lds_addr), "v"((unsigned long long)gaddr) : "memory");
}

// ---------------------------------------------------------------------------
// Kernel 1: f32 -> bf16 conversion, 8 elements / thread (b128 streams)
// ---------------------------------------------------------------------------
__global__ void cvt_f32_bf16_x8(const float* __restrict__ src,
                                unsigned short* __restrict__ dst, int n8) {
    int i = blockIdx.x * blockDim.x + threadIdx.x;
    if (i >= n8) return;
    const float4* s = (const float4*)(src) + 2 * (size_t)i;
    float4 a = s[0], b = s[1];
    uint4 w;
    w.x = (unsigned)f2bf(a.x) | ((unsigned)f2bf(a.y) << 16);
    w.y = (unsigned)f2bf(a.z) | ((unsigned)f2bf(a.w) << 16);
    w.z = (unsigned)f2bf(b.x) | ((unsigned)f2bf(b.y) << 16);
    w.w = (unsigned)f2bf(b.z) | ((unsigned)f2bf(b.w) << 16);
    *((uint4*)(dst) + i) = w;
}

// ---------------------------------------------------------------------------
// Kernel 2: QKV projection GEMM (A = xb [M,K=D], W rows over d, both bf16).
// 128x128 C tile / block (256 threads, 8 waves).  Wave: 32 rows x 64 cols,
// 8 WMMA per k-step.  Double-buffered async-to-LDS tile pipeline.
// vmode=0: out [B][H][T][Dh] (Q,K);  vmode=1: out [B][H][Dh][T] (V, packed).
// ---------------------------------------------------------------------------
__global__ __launch_bounds__(256) void qkv_gemm(
    const unsigned short* __restrict__ xb,
    const unsigned short* __restrict__ wb,
    const float* __restrict__ bias,
    unsigned short* __restrict__ out, int vmode) {
    __shared__ unsigned short Ab[2][128][40];
    __shared__ unsigned short Bb[2][128][40];

    const int tid  = threadIdx.x;
    const int lane = tid & 31;
    const int wave = tid >> 5;
    const int half = lane >> 4;
    const int idx  = lane & 15;
    const int wr   = wave & 3;   // 4 row positions of 32
    const int wc   = wave >> 2;  // 2 col positions of 64
    const int m0   = blockIdx.x * 128;
    const int n0   = blockIdx.y * 128;

    // async-load assignment: 128 rows x 4 segs(16B) per matrix, 2 per thread
    const int r0  = tid >> 2;
    const int r1  = 64 + (tid >> 2);
    const int off = (tid & 3) * 8;   // shorts

    auto issue = [&](int kt) {
        const int buf = kt & 1;
        async_b128((unsigned)(size_t)&Ab[buf][r0][off],
                   xb + (size_t)(m0 + r0) * CD + kt * 32 + off);
        async_b128((unsigned)(size_t)&Ab[buf][r1][off],
                   xb + (size_t)(m0 + r1) * CD + kt * 32 + off);
        async_b128((unsigned)(size_t)&Bb[buf][r0][off],
                   wb + (size_t)(n0 + r0) * CD + kt * 32 + off);
        async_b128((unsigned)(size_t)&Bb[buf][r1][off],
                   wb + (size_t)(n0 + r1) * CD + kt * 32 + off);
    };

    Acc8 acc[2][4];
#pragma unroll
    for (int u = 0; u < 2; u++)
#pragma unroll
        for (int t = 0; t < 4; t++)
#pragma unroll
            for (int v = 0; v < 8; v++) acc[u][t].f[v] = 0.0f;

    issue(0);
    issue(1);

    for (int kt = 0; kt < KTILES; kt++) {
        const int buf = kt & 1;
        if (kt + 1 < KTILES)
            asm volatile("s_wait_asynccnt 4" ::: "memory");  // current tile landed
        else
            asm volatile("s_wait_asynccnt 0" ::: "memory");
        __syncthreads();

        Frag16 fa[2], fb[4];
#pragma unroll
        for (int u = 0; u < 2; u++) {
            const int am = wr * 32 + u * 16 + idx;
#pragma unroll
            for (int j = 0; j < 8; j++)
                fa[u].d[j] = *(const unsigned int*)&Ab[buf][am][a_pair_k(j, half)];
        }
#pragma unroll
        for (int t = 0; t < 4; t++) {
            const int bn = wc * 64 + t * 16 + idx;
#pragma unroll
            for (int j = 0; j < 8; j++)
                fb[t].d[j] = *(const unsigned int*)&Bb[buf][bn][a_pair_k(j, half)];
        }
#pragma unroll
        for (int t = 0; t < 4; t++)
#pragma unroll
            for (int u = 0; u < 2; u++)
                acc[u][t].v = __builtin_amdgcn_wmma_f32_16x16x32_bf16(
                    false, fa[u].v, false, fb[t].v, (short)0, acc[u][t].v, false, false);

        __syncthreads();                     // everyone done reading buf
        if (kt + 2 < KTILES) issue(kt + 2);  // refill freed buffer
    }

    // epilogue: add bias
#pragma unroll
    for (int t = 0; t < 4; t++) {
        const int n  = n0 + wc * 64 + t * 16 + idx;
        const int h  = n >> 6;    // Dh = 64
        const int kh = n & 63;
        const float bv = bias[n];
#pragma unroll
        for (int u = 0; u < 2; u++) {
            const int mbase = m0 + wr * 32 + u * 16 + half * 8;
            const int bb    = mbase >> 11;    // T = 2048
            const int tok0  = mbase & 2047;
            if (vmode == 0) {
                // [B][H][T][Dh]
#pragma unroll
                for (int v = 0; v < 8; v++) {
                    size_t o = (((size_t)(bb * CH + h)) * CT + tok0 + v) * CDH + kh;
                    out[o] = f2bf(acc[u][t].f[v] + bv);
                }
            } else {
                // [B][H][Dh][T]: 8 consecutive tokens -> one 16B packed store
                unsigned short hh[8];
#pragma unroll
                for (int v = 0; v < 8; v++) hh[v] = f2bf(acc[u][t].f[v] + bv);
                uint4 w;
                w.x = (unsigned)hh[0] | ((unsigned)hh[1] << 16);
                w.y = (unsigned)hh[2] | ((unsigned)hh[3] << 16);
                w.z = (unsigned)hh[4] | ((unsigned)hh[5] << 16);
                w.w = (unsigned)hh[6] | ((unsigned)hh[7] << 16);
                size_t o = (((size_t)(bb * CH + h)) * CDH + kh) * CT + tok0;
                *(uint4*)&out[o] = w;
            }
        }
    }
}

// ---------------------------------------------------------------------------
// Kernel 3: flash attention over one (b,h), 128 q-rows per block.
// 8 waves x 16 q-rows; K/V tiles shared through LDS via double-buffered
// async-to-LDS pipeline (2 async b128 / thread / s-step); online softmax;
// P restaged via LDS for P*V WMMA.
// Q/K: [B][H][T][Dh] bf16.  Vt: [B][H][Dh][T] bf16.  outh: [B][T][H][Dh] bf16.
// ---------------------------------------------------------------------------
__global__ __launch_bounds__(256) void attn_kernel(
    const unsigned short* __restrict__ Q,
    const unsigned short* __restrict__ K,
    const unsigned short* __restrict__ Vt,
    unsigned short* __restrict__ outh) {
    __shared__ unsigned short Kt[2][32][72];  // 144B rows: 16B-aligned, conflict-free
    __shared__ unsigned short Vs[2][64][40];  // 80B rows
    __shared__ unsigned short Pl[8][16][34];

    const int tid  = threadIdx.x;
    const int lane = tid & 31;
    const int wave = tid >> 5;
    const int half = lane >> 4;
    const int idx  = lane & 15;

    const int bh = blockIdx.y;
    const int b  = bh >> 4;   // H = 16
    const int h  = bh & 15;
    const size_t base = (size_t)bh * CT * CDH;   // same total size for Vt
    const int q0 = blockIdx.x * 128 + wave * 16;

    // async seg assignment: K tile 32x64 (8 segs/row), V tile 64x32 (4 segs/row)
    const int krow = tid >> 3, kcol = (tid & 7) * 8;
    const int vrow = tid >> 2, vcol = (tid & 3) * 8;

    auto issue = [&](int it) {
        const int buf = it & 1;
        async_b128((unsigned)(size_t)&Kt[buf][krow][kcol],
                   K + base + (size_t)(it * 32 + krow) * CDH + kcol);
        async_b128((unsigned)(size_t)&Vs[buf][vrow][vcol],
                   Vt + base + (size_t)vrow * CT + it * 32 + vcol);
    };

    // Q fragments for this wave's 16 rows (2 k-chunks of 32 over Dh=64)
    Frag16 qa[2];
    {
        const size_t rowp = base + (size_t)(q0 + idx) * CDH;
#pragma unroll
        for (int kc = 0; kc < 2; kc++)
#pragma unroll
            for (int j = 0; j < 8; j++)
                qa[kc].d[j] =
                    *(const unsigned int*)(Q + rowp + kc * 32 + a_pair_k(j, half));
    }

    Acc8 o[4];
    float mrow[8], lrow[8];
#pragma unroll
    for (int t = 0; t < 4; t++)
#pragma unroll
        for (int v = 0; v < 8; v++) o[t].f[v] = 0.0f;
#pragma unroll
    for (int v = 0; v < 8; v++) { mrow[v] = -1e30f; lrow[v] = 0.0f; }

    issue(0);
    issue(1);

    for (int it = 0; it < STILES; it++) {
        const int buf = it & 1;
        if (it + 1 < STILES)
            asm volatile("s_wait_asynccnt 2" ::: "memory");  // current tiles landed
        else
            asm volatile("s_wait_asynccnt 0" ::: "memory");
        __syncthreads();

        // ---- K and V B-fragments from LDS (b128 reads, overlapable) ----
        Frag16 kb[2][2], vb[4];
#pragma unroll
        for (int t = 0; t < 2; t++)
#pragma unroll
            for (int kc = 0; kc < 2; kc++)
#pragma unroll
                for (int j = 0; j < 8; j++)
                    kb[t][kc].d[j] = *(const unsigned int*)
                        &Kt[buf][t * 16 + idx][kc * 32 + a_pair_k(j, half)];
#pragma unroll
        for (int t = 0; t < 4; t++)
#pragma unroll
            for (int j = 0; j < 8; j++)
                vb[t].d[j] = *(const unsigned int*)
                    &Vs[buf][t * 16 + idx][a_pair_k(j, half)];

        // ---- scores S = Q * K^T for 16 q-rows x 32 s-cols (2 C frags) ----
        Acc8 sc[2];
#pragma unroll
        for (int t = 0; t < 2; t++) {
#pragma unroll
            for (int v = 0; v < 8; v++) sc[t].f[v] = 0.0f;
#pragma unroll
            for (int kc = 0; kc < 2; kc++)
                sc[t].v = __builtin_amdgcn_wmma_f32_16x16x32_bf16(
                    false, qa[kc].v, false, kb[t][kc].v, (short)0, sc[t].v, false, false);
        }

        // ---- online softmax (rows m = half*8+v live across 16 lanes of half) ----
        const float scale = 0.125f;  // 1/sqrt(64)
#pragma unroll
        for (int v = 0; v < 8; v++) {
            float x0 = sc[0].f[v] * scale;
            float x1 = sc[1].f[v] * scale;
            float mx = fmaxf(x0, x1);
#pragma unroll
            for (int d = 1; d < 16; d <<= 1) mx = fmaxf(mx, __shfl_xor(mx, d, 32));
            float mn    = fmaxf(mrow[v], mx);
            float alpha = __expf(mrow[v] - mn);
            mrow[v]     = mn;
            float p0 = __expf(x0 - mn);
            float p1 = __expf(x1 - mn);
            sc[0].f[v] = p0;
            sc[1].f[v] = p1;
            float rs = p0 + p1;
#pragma unroll
            for (int d = 1; d < 16; d <<= 1) rs += __shfl_xor(rs, d, 32);
            lrow[v] = lrow[v] * alpha + rs;
#pragma unroll
            for (int t = 0; t < 4; t++) o[t].f[v] *= alpha;
        }

        // ---- stage P (bf16) through LDS: C-layout -> row-major [16][32] ----
#pragma unroll
        for (int t = 0; t < 2; t++)
#pragma unroll
            for (int v = 0; v < 8; v++)
                Pl[wave][half * 8 + v][t * 16 + idx] = f2bf(sc[t].f[v]);
        asm volatile("s_wait_dscnt 0" ::: "memory");  // same-wave LDS RAW fence

        Frag16 pa;
#pragma unroll
        for (int j = 0; j < 8; j++)
            pa.d[j] = *(const unsigned int*)&Pl[wave][idx][a_pair_k(j, half)];

        // ---- O += P * V (4 d-tiles of 16) ----
#pragma unroll
        for (int t = 0; t < 4; t++)
            o[t].v = __builtin_amdgcn_wmma_f32_16x16x32_bf16(
                false, pa.v, false, vb[t].v, (short)0, o[t].v, false, false);

        __syncthreads();                     // all waves done with buf
        if (it + 2 < STILES) issue(it + 2);  // refill freed buffer
    }

    // ---- finalize: O /= l, store to [B][T][H][Dh] bf16 ----
#pragma unroll
    for (int v = 0; v < 8; v++) {
        float inv = 1.0f / lrow[v];
        int q = q0 + half * 8 + v;
#pragma unroll
        for (int t = 0; t < 4; t++) {
            int dcol = t * 16 + idx;
            outh[(((size_t)b * CT + q) * CH + h) * CDH + dcol] =
                f2bf(o[t].f[v] * inv);
        }
    }
}

// ---------------------------------------------------------------------------
// Kernel 4: output projection, same 128x128 async-pipelined GEMM.
//   A  = outh [M=B*T, K=H*Dh=1024] bf16;  W = WOb [H][D][Dh] bf16.
//   out f32 [B*T, D] + bias.
// ---------------------------------------------------------------------------
__global__ __launch_bounds__(256) void oproj_gemm(
    const unsigned short* __restrict__ A,
    const unsigned short* __restrict__ Wb,
    const float* __restrict__ bO,
    float* __restrict__ out) {
    __shared__ unsigned short Ab[2][128][40];
    __shared__ unsigned short Bb[2][128][40];

    const int tid  = threadIdx.x;
    const int lane = tid & 31;
    const int wave = tid >> 5;
    const int half = lane >> 4;
    const int idx  = lane & 15;
    const int wr   = wave & 3;
    const int wc   = wave >> 2;
    const int m0   = blockIdx.x * 128;
    const int n0   = blockIdx.y * 128;

    const int r0  = tid >> 2;
    const int r1  = 64 + (tid >> 2);
    const int off = (tid & 3) * 8;

    auto issue = [&](int kt) {
        const int buf  = kt & 1;
        const int hh   = kt >> 1;         // k0>>6 with k0 = kt*32
        const int koff = (kt & 1) * 32;   // k0&63
        async_b128((unsigned)(size_t)&Ab[buf][r0][off],
                   A + (size_t)(m0 + r0) * CD + kt * 32 + off);
        async_b128((unsigned)(size_t)&Ab[buf][r1][off],
                   A + (size_t)(m0 + r1) * CD + kt * 32 + off);
        async_b128((unsigned)(size_t)&Bb[buf][r0][off],
                   Wb + ((size_t)hh * CD + (n0 + r0)) * CDH + koff + off);
        async_b128((unsigned)(size_t)&Bb[buf][r1][off],
                   Wb + ((size_t)hh * CD + (n0 + r1)) * CDH + koff + off);
    };

    Acc8 acc[2][4];
#pragma unroll
    for (int u = 0; u < 2; u++)
#pragma unroll
        for (int t = 0; t < 4; t++)
#pragma unroll
            for (int v = 0; v < 8; v++) acc[u][t].f[v] = 0.0f;

    issue(0);
    issue(1);

    for (int kt = 0; kt < KTILES; kt++) {
        const int buf = kt & 1;
        if (kt + 1 < KTILES)
            asm volatile("s_wait_asynccnt 4" ::: "memory");
        else
            asm volatile("s_wait_asynccnt 0" ::: "memory");
        __syncthreads();

        Frag16 fa[2], fb[4];
#pragma unroll
        for (int u = 0; u < 2; u++) {
            const int am = wr * 32 + u * 16 + idx;
#pragma unroll
            for (int j = 0; j < 8; j++)
                fa[u].d[j] = *(const unsigned int*)&Ab[buf][am][a_pair_k(j, half)];
        }
#pragma unroll
        for (int t = 0; t < 4; t++) {
            const int bn = wc * 64 + t * 16 + idx;
#pragma unroll
            for (int j = 0; j < 8; j++)
                fb[t].d[j] = *(const unsigned int*)&Bb[buf][bn][a_pair_k(j, half)];
        }
#pragma unroll
        for (int t = 0; t < 4; t++)
#pragma unroll
            for (int u = 0; u < 2; u++)
                acc[u][t].v = __builtin_amdgcn_wmma_f32_16x16x32_bf16(
                    false, fa[u].v, false, fb[t].v, (short)0, acc[u][t].v, false, false);

        __syncthreads();
        if (kt + 2 < KTILES) issue(kt + 2);
    }

#pragma unroll
    for (int t = 0; t < 4; t++) {
        const int n = n0 + wc * 64 + t * 16 + idx;
        const float bv = bO[n];
#pragma unroll
        for (int u = 0; u < 2; u++)
#pragma unroll
            for (int v = 0; v < 8; v++) {
                int m = m0 + wr * 32 + u * 16 + half * 8 + v;
                out[(size_t)m * CD + n] = acc[u][t].f[v] + bv;
            }
    }
}

// ---------------------------------------------------------------------------
// Host launcher
// ---------------------------------------------------------------------------
extern "C" void kernel_launch(void* const* d_in, const int* in_sizes, int n_in,
                              void* d_out, int out_size, void* d_ws, size_t ws_size,
                              hipStream_t stream) {
    (void)in_sizes; (void)n_in; (void)out_size; (void)ws_size;

    const float* x   = (const float*)d_in[0];
    const float* W_Q = (const float*)d_in[1];
    const float* b_Q = (const float*)d_in[2];
    const float* W_K = (const float*)d_in[3];
    const float* b_K = (const float*)d_in[4];
    const float* W_V = (const float*)d_in[5];
    const float* b_V = (const float*)d_in[6];
    const float* W_O = (const float*)d_in[7];
    const float* b_O = (const float*)d_in[8];
    float* out = (float*)d_out;

    const size_t nx = (size_t)CB * CT * CD;   // 8,388,608
    const size_t nw = (size_t)CH * CDH * CD;  // 1,048,576
    const size_t nq = (size_t)CB * CH * CT * CDH;

    char* ws = (char*)d_ws;
    unsigned short* xb  = (unsigned short*)ws; ws += nx * 2;
    unsigned short* wqb = (unsigned short*)ws; ws += nw * 2;
    unsigned short* wkb = (unsigned short*)ws; ws += nw * 2;
    unsigned short* wvb = (unsigned short*)ws; ws += nw * 2;
    unsigned short* wob = (unsigned short*)ws; ws += nw * 2;
    unsigned short* Qb  = (unsigned short*)ws; ws += nq * 2;
    unsigned short* Kb  = (unsigned short*)ws; ws += nq * 2;
    unsigned short* Vtb = (unsigned short*)ws; ws += nq * 2;  // [B][H][Dh][T]
    unsigned short* Ob  = (unsigned short*)ws; ws += nq * 2;

    // 1) bf16 conversions (8 elems/thread, all sizes divisible by 8*256)
    cvt_f32_bf16_x8<<<(int)(nx / 8 / 256), 256, 0, stream>>>(x, xb, (int)(nx / 8));
    cvt_f32_bf16_x8<<<(int)(nw / 8 / 256), 256, 0, stream>>>(W_Q, wqb, (int)(nw / 8));
    cvt_f32_bf16_x8<<<(int)(nw / 8 / 256), 256, 0, stream>>>(W_K, wkb, (int)(nw / 8));
    cvt_f32_bf16_x8<<<(int)(nw / 8 / 256), 256, 0, stream>>>(W_V, wvb, (int)(nw / 8));
    cvt_f32_bf16_x8<<<(int)(nw / 8 / 256), 256, 0, stream>>>(W_O, wob, (int)(nw / 8));

    // 2) QKV projections: M=8192 (64 tiles), N=1024 (8 tiles)
    dim3 gg(64, 8);
    qkv_gemm<<<gg, 256, 0, stream>>>(xb, wqb, b_Q, Qb, 0);
    qkv_gemm<<<gg, 256, 0, stream>>>(xb, wkb, b_K, Kb, 0);
    qkv_gemm<<<gg, 256, 0, stream>>>(xb, wvb, b_V, Vtb, 1);  // V transposed

    // 3) attention: (T/128, B*H) blocks of 256 threads
    dim3 ga(CT / 128, CB * CH);
    attn_kernel<<<ga, 256, 0, stream>>>(Qb, Kb, Vtb, Ob);

    // 4) output projection -> f32 result
    oproj_gemm<<<gg, 256, 0, stream>>>(Ob, wob, b_O, out);
}
